// MaskedSelfAttention_1254130450638
// MI455X (gfx1250) — matrix operand order repaired and checked by reading, not compiled, not run on hardware
//
#include <hip/hip_runtime.h>
#include <hip/hip_bf16.h>
#include <cstdint>

// ---------------------------------------------------------------------------
// MaskedSelfAttention on MI455X (gfx1250), wave32 + V_WMMA_F32_16X16X32_F16.
// Pipeline: cvt(f32->f16) -> WMMA GEMM (QKV) -> flash-attn (WMMA + async LDS)
//           -> WMMA GEMM (out proj, f32 out).
// ---------------------------------------------------------------------------

typedef _Float16 v16h __attribute__((ext_vector_type(16)));
typedef float    v8f  __attribute__((ext_vector_type(8)));

union H16 {
    v16h      v;
    unsigned  u[8];
    _Float16  h[16];
};

// A-operand (16x32 f16) K-offset for VGPR v, half-wave hi:
//   lanes 0-15 : V0..3 -> K=0..7,  V4..7 -> K=16..23
//   lanes 16-31: V0..3 -> K=8..15, V4..7 -> K=24..31
__device__ __forceinline__ int koffA(int v, int hi) {
    int base = (v < 4) ? (2 * v) : (16 + 2 * (v - 4));
    return base + (hi ? 8 : 0);
}
// B-operand (32x16 f16): lane = column, same K mapping as A (koffA)
__device__ __forceinline__ int koffB(int v, int hi) {
    return koffA(v, hi);
}

__device__ __forceinline__ v8f wmma32(const v16h& a, const v16h& b, v8f c) {
    return __builtin_amdgcn_wmma_f32_16x16x32_f16(
        /*neg_a=*/false, a, /*neg_b=*/false, b,
        /*c_mod=*/(short)0, c, /*reuse_a=*/false, /*reuse_b=*/false);
}

// Low 32 bits of a flat LDS pointer == workgroup LDS byte offset (LDS aperture
// is 4GB aligned).
__device__ __forceinline__ unsigned lds_byte_off(const void* p) {
    return (unsigned)(uintptr_t)p;
}

// Async-copy 4KB (contiguous) global -> LDS: 8 x b128 per lane (32 lanes*16B).
__device__ __forceinline__ void async_copy_4k(const _Float16* g, _Float16* s, int lane) {
    unsigned lbase = lds_byte_off(s);
    unsigned long long gbase = (unsigned long long)(uintptr_t)g;
#pragma unroll
    for (int i = 0; i < 8; ++i) {
        unsigned loff = lbase + (unsigned)(lane * 16 + i * 512);
        unsigned long long ga = gbase + (unsigned long long)(lane * 16 + i * 512);
        asm volatile("global_load_async_to_lds_b128 %0, %1, off"
                     :: "v"(loff), "v"(ga) : "memory");
    }
}

// ---------------------------------------------------------------------------
// f32 -> f16 elementwise conversion
// ---------------------------------------------------------------------------
__global__ void cvt_f32_f16(const float* __restrict__ in, _Float16* __restrict__ out, int n) {
    int i = blockIdx.x * blockDim.x + threadIdx.x;
    if (i < n) out[i] = (_Float16)in[i];
}

// ---------------------------------------------------------------------------
// GEMM: Cf32[m][n] = sum_k A[m][k] * B[n][k] + bias[n]        (A@B^T + bias)
// One wave computes a 16x64 output tile (4 WMMAs per K-step, A reused).
// mode 0: write f32 to Cout[m*N+n]
// mode 1: QKV scatter -> q16/k16/v16 as f16 in [B*H][T][64]
// ---------------------------------------------------------------------------
__global__ __launch_bounds__(32)
void gemm_wmma(const _Float16* __restrict__ A, const _Float16* __restrict__ Bm,
               const float* __restrict__ bias, float* __restrict__ Cout,
               _Float16* __restrict__ q16, _Float16* __restrict__ k16,
               _Float16* __restrict__ v16,
               int M, int N, int K, int mode, int T)
{
    const int lane = threadIdx.x & 31;
    const int hi   = lane >> 4;
    const int r    = lane & 15;
    const int m0   = blockIdx.x * 16;
    const int n0   = blockIdx.y * 64;

    v8f zero = {};
    v8f acc[4] = {zero, zero, zero, zero};

    const _Float16* arow = A + (size_t)(m0 + r) * K;

    for (int k0 = 0; k0 < K; k0 += 32) {
        __builtin_prefetch(arow + k0 + 256, 0, 1);
        H16 a;
#pragma unroll
        for (int v = 0; v < 8; ++v)
            a.u[v] = *(const unsigned*)(arow + k0 + koffA(v, hi));
#pragma unroll
        for (int nt = 0; nt < 4; ++nt) {
            const _Float16* brow = Bm + (size_t)(n0 + nt * 16 + r) * K;
            H16 b;
#pragma unroll
            for (int v = 0; v < 8; ++v)
                b.u[v] = *(const unsigned*)(brow + k0 + koffB(v, hi));
            acc[nt] = wmma32(a.v, b.v, acc[nt]);
        }
    }

#pragma unroll
    for (int nt = 0; nt < 4; ++nt) {
        const int gn = n0 + nt * 16 + r;        // column
        const float bv = bias[gn];
#pragma unroll
        for (int i = 0; i < 8; ++i) {
            const int gm = m0 + i + 8 * hi;     // row
            float val = acc[nt][i] + bv;
            if (mode == 0) {
                Cout[(size_t)gm * N + gn] = val;
            } else {
                const int C1 = N / 3;           // = C = 1024
                const int which = gn / C1;
                const int cc = gn % C1;
                const int h = cc >> 6;
                const int d = cc & 63;
                const int b = gm / T, t = gm % T;
                const size_t idx = (((size_t)(b * 16 + h)) * T + t) * 64 + d;
                _Float16 hv = (_Float16)val;
                if (which == 0)      q16[idx] = hv;
                else if (which == 1) k16[idx] = hv;
                else                 v16[idx] = hv;
            }
        }
    }
}

// ---------------------------------------------------------------------------
// Flash attention: one wave per (bh, 16-query tile). Causal softmax online.
// Q/K/V are f16 [B*H][T][64]. Output Y f16 [B][T][H*64].
// ---------------------------------------------------------------------------
__global__ __launch_bounds__(32)
void attn_wmma(const _Float16* __restrict__ Qm, const _Float16* __restrict__ Km,
               const _Float16* __restrict__ Vm, _Float16* __restrict__ Y, int T)
{
    __shared__ _Float16 sK[32 * 64];   // [key][d]
    __shared__ _Float16 sV[32 * 64];   // [key][d]
    __shared__ _Float16 sQ[16 * 64];   // [q][d]
    __shared__ _Float16 sP[16 * 32];   // [q][key]

    const int lane = threadIdx.x & 31;
    const int hi   = lane >> 4;
    const int r    = lane & 15;
    const int bh   = blockIdx.y;
    const int qb   = blockIdx.x * 16;

    // stage Q tile (2KB) with coalesced 16B loads
    {
        const uint4* src = (const uint4*)(Qm + ((size_t)bh * T + qb) * 64);
        uint4* dst = (uint4*)sQ;
#pragma unroll
        for (int i = lane; i < (16 * 64 * 2) / 16; i += 32) dst[i] = src[i];
    }
    __syncthreads();

    H16 aQ[2];
#pragma unroll
    for (int c = 0; c < 2; ++c)
#pragma unroll
        for (int v = 0; v < 8; ++v)
            aQ[c].u[v] = *(const unsigned*)&sQ[r * 64 + c * 32 + koffA(v, hi)];

    v8f zero = {};
    v8f o[4] = {zero, zero, zero, zero};
    float mrow[8], lrow[8];
#pragma unroll
    for (int i = 0; i < 8; ++i) { mrow[i] = -3.0e38f; lrow[i] = 0.0f; }

    const int nkb = (qb + 16 + 31) >> 5;     // key blocks of 32 (causal bound)
    for (int kb = 0; kb < nkb; ++kb) {
        const int kbase = kb << 5;
        async_copy_4k(Km + ((size_t)bh * T + kbase) * 64, sK, lane);
        async_copy_4k(Vm + ((size_t)bh * T + kbase) * 64, sV, lane);
        asm volatile("s_wait_asynccnt 0x0" ::: "memory");
        __syncthreads();

        // S = Q * K^T (two 16x16 key tiles, K-dim = head dim 64 in 2 steps)
        float sv[2][8];
#pragma unroll
        for (int tile = 0; tile < 2; ++tile) {
            H16 bK[2];
#pragma unroll
            for (int c = 0; c < 2; ++c)
#pragma unroll
                for (int v = 0; v < 8; ++v)
                    bK[c].u[v] = *(const unsigned*)
                        &sK[(tile * 16 + r) * 64 + c * 32 + koffB(v, hi)];
            v8f st = zero;
            st = wmma32(aQ[0].v, bK[0].v, st);
            st = wmma32(aQ[1].v, bK[1].v, st);
#pragma unroll
            for (int i = 0; i < 8; ++i) {
                const int q  = qb + i + 8 * hi;
                const int kg = kbase + tile * 16 + r;
                float x = st[i] * 0.125f;                 // 1/sqrt(64)
                sv[tile][i] = (kg <= q) ? x : -3.0e38f;   // causal mask
            }
        }

        // online softmax: row stats via 16-lane shfl_xor reductions
        float esc[8];
#pragma unroll
        for (int i = 0; i < 8; ++i) {
            float mx = fmaxf(sv[0][i], sv[1][i]);
            for (int off = 1; off < 16; off <<= 1)
                mx = fmaxf(mx, __shfl_xor(mx, off, 32));
            const float mnew = fmaxf(mrow[i], mx);
            esc[i] = __expf(mrow[i] - mnew);
            float p0 = __expf(sv[0][i] - mnew);
            float p1 = __expf(sv[1][i] - mnew);
            sv[0][i] = p0; sv[1][i] = p1;
            float rs = p0 + p1;
            for (int off = 1; off < 16; off <<= 1)
                rs += __shfl_xor(rs, off, 32);
            lrow[i] = lrow[i] * esc[i] + rs;
            mrow[i] = mnew;
        }
#pragma unroll
        for (int t = 0; t < 4; ++t)
#pragma unroll
            for (int i = 0; i < 8; ++i)
                o[t][i] *= esc[i];

        // P: C-layout -> LDS -> A-layout for P*V
#pragma unroll
        for (int tile = 0; tile < 2; ++tile)
#pragma unroll
            for (int i = 0; i < 8; ++i)
                sP[(i + 8 * hi) * 32 + tile * 16 + r] = (_Float16)sv[tile][i];
        __syncthreads();

        H16 aP;
#pragma unroll
        for (int v = 0; v < 8; ++v)
            aP.u[v] = *(const unsigned*)&sP[r * 32 + koffA(v, hi)];

#pragma unroll
        for (int t = 0; t < 4; ++t) {
            H16 bV;   // B element (k=key, n=d): gather from sV[key][d]
#pragma unroll
            for (int v = 0; v < 8; ++v) {
                const int k0 = koffB(v, hi);
                bV.h[2 * v]     = sV[(k0)     * 64 + t * 16 + r];
                bV.h[2 * v + 1] = sV[(k0 + 1) * 64 + t * 16 + r];
            }
            o[t] = wmma32(aP.v, bV.v, o[t]);
        }
        __syncthreads();
    }

    // epilogue: Y[b][row][h*64+d] = o / l
    const int b = bh >> 4, h = bh & 15;
#pragma unroll
    for (int t = 0; t < 4; ++t)
#pragma unroll
        for (int i = 0; i < 8; ++i) {
            const int row = qb + i + 8 * hi;
            const int d   = t * 16 + r;
            const float val = o[t][i] / lrow[i];
            Y[((size_t)b * T + row) * 1024 + h * 64 + d] = (_Float16)val;
        }
}

// ---------------------------------------------------------------------------
// Host launcher
// ---------------------------------------------------------------------------
extern "C" void kernel_launch(void* const* d_in, const int* in_sizes, int n_in,
                              void* d_out, int out_size, void* d_ws, size_t ws_size,
                              hipStream_t stream) {
    const float* x  = (const float*)d_in[0];
    const float* We = (const float*)d_in[1];
    const float* be = (const float*)d_in[2];
    const float* Wp = (const float*)d_in[3];
    const float* bp = (const float*)d_in[4];

    const int B = 2, T = 2048, C = 1024, H = 16;
    const int M = B * T;                 // 4096

    char* ws = (char*)d_ws;
    _Float16* x16  = (_Float16*)ws; ws += (size_t)M * C * 2;
    _Float16* We16 = (_Float16*)ws; ws += (size_t)3 * C * C * 2;
    _Float16* Wp16 = (_Float16*)ws; ws += (size_t)C * C * 2;
    _Float16* q16  = (_Float16*)ws; ws += (size_t)B * H * T * 64 * 2;
    _Float16* k16  = (_Float16*)ws; ws += (size_t)B * H * T * 64 * 2;
    _Float16* v16  = (_Float16*)ws; ws += (size_t)B * H * T * 64 * 2;
    _Float16* y16  = (_Float16*)ws; ws += (size_t)M * C * 2;

    const int nx = M * C, nwe = 3 * C * C, nwp = C * C;
    cvt_f32_f16<<<(nx  + 255) / 256, 256, 0, stream>>>(x,  x16,  nx);
    cvt_f32_f16<<<(nwe + 255) / 256, 256, 0, stream>>>(We, We16, nwe);
    cvt_f32_f16<<<(nwp + 255) / 256, 256, 0, stream>>>(Wp, Wp16, nwp);

    // qkv = x @ We^T + be  (scatter into q/k/v head-major f16)
    gemm_wmma<<<dim3(M / 16, (3 * C) / 64), 32, 0, stream>>>(
        x16, We16, be, nullptr, q16, k16, v16, M, 3 * C, C, /*mode=*/1, T);

    // flash attention
    attn_wmma<<<dim3(T / 16, B * H), 32, 0, stream>>>(q16, k16, v16, y16, T);

    // out = y @ Wp^T + bp  (f32)
    gemm_wmma<<<dim3(M / 16, C / 64), 32, 0, stream>>>(
        y16, Wp16, bp, (float*)d_out, nullptr, nullptr, nullptr,
        M, C, C, /*mode=*/0, T);
}
